// PromptAdapater_39994735460539
// MI455X (gfx1250) — matrix operand order repaired and checked by reading, not compiled, not run on hardware
//
#include <hip/hip_runtime.h>
#include <hip/hip_bf16.h>
#include <cstddef>

// ---------------------------------------------------------------------------
// Sizes (match reference)
// ---------------------------------------------------------------------------
#define BB   32
#define TT   8
#define CC   256
#define HW   64
#define NP   4096          // HW*HW
#define NH   8
#define DH   16            // CC/2 / NH
#define INT_ 128           // internal dim
#define HID  64

#define USE_ASYNC_LDS 1

typedef __attribute__((ext_vector_type(16))) __bf16    v16bf;
typedef __attribute__((ext_vector_type(8)))  float     v8f;
typedef __attribute__((ext_vector_type(4)))  unsigned  uint4v;

union FragBF { v16bf v; uint4v q[2]; unsigned u[8]; };

__device__ __forceinline__ unsigned short f2bf(float f) {
    union { float f; unsigned u; } x; x.f = f;
    unsigned r = x.u + 0x7FFFu + ((x.u >> 16) & 1u);   // round-to-nearest-even
    return (unsigned short)(r >> 16);
}
__device__ __forceinline__ float bf_lo(unsigned u) {
    union { unsigned u; float f; } x; x.u = u << 16; return x.f;
}
__device__ __forceinline__ float bf_hi(unsigned u) {
    union { unsigned u; float f; } x; x.u = u & 0xFFFF0000u; return x.f;
}

// ---------------------------------------------------------------------------
// K1 LDS map (dynamic LDS; kernel has no static __shared__, so the dynamic
// region starts at LDS offset 0 — raw byte offsets usable by async-to-LDS):
//   [0,      65792)  ldsG  f32 [64][257]                 (ge tile, transposed)
//   [65792, 98560)   A_k   frag-major [4 prow][8 s][32 lane][32B]
//   [98560, 131328)  A_v   frag-major
//   [131328,196864)  WkT   frag-major [8 jt][8 s][32 lane][32B]
//   [196864,262400)  WvT   frag-major
// ---------------------------------------------------------------------------
#define OFF_AK 65792
#define OFF_AV 98560
#define OFF_WK 131328
#define OFF_WV 196864
#define K1_LDS 262400

// A-operand (16x32 bf16) fragment-major byte offset for element (p, c):
// lanes 0-15 hold M=lane, K=0..15; lanes 16-31 hold M=lane-16,
// K=16..31; VGPR i -> K pair (kb+2i, kb+2i+1).
__device__ __forceinline__ unsigned a_swz(int p, int c) {
    const int prow = p >> 4, m = p & 15, s = c >> 5, kk = c & 31;
    const int lh   = kk >> 4;
    const int lane = m + (lh << 4);
    const int pi   = (kk & 15) >> 1;
    return (unsigned)((((prow * 8 + s) * 32 + lane) << 5) + (pi << 2) + ((kk & 1) << 1));
}

// B-operand (32x16 bf16) fragment-major byte offset for element (k-row c, col j):
// lanes 0-15 hold N=lane, K=0..15; lanes 16-31 hold N=lane-16, K=16..31;
// VGPR r -> K pair (kb+2r, kb+2r+1).
__device__ __forceinline__ unsigned b_swz(int c, int j) {
    const int jt = j >> 4, n = j & 15, s = c >> 5, kk = c & 31;
    const int lane = n + ((kk >> 4) << 4);
    const int r    = (kk & 15) >> 1;
    return (unsigned)((((jt * 8 + s) * 32 + lane) << 5) + (r << 2) + ((kk & 1) << 1));
}

// ---------------------------------------------------------------------------
// K1: fused  guiding = ge^T * keys ;  K = (guiding+key_pe)@wk+bk ; V = guiding@wv+bv
// WMMA bf16 16x16x32, f32 accumulate.  One WG = (batch, 64-position tile).
// ---------------------------------------------------------------------------
__global__ __launch_bounds__(512)
void k1_guiding_kvproj(const float* __restrict__ ge,      // (B,C,4096)
                       const float* __restrict__ keys,    // (B,4096,C)
                       const float* __restrict__ key_pe,  // (B,4096,C)
                       const float* __restrict__ wk, const float* __restrict__ bk,
                       const float* __restrict__ wv, const float* __restrict__ bv,
                       float* __restrict__ guiding_out,          // (B,4096,C)
                       unsigned short* __restrict__ k_ws,        // bf16 (B,4096,128)
                       unsigned short* __restrict__ v_ws)        // bf16 (B,4096,128)
{
    extern __shared__ char smem[];
    float* ldsG = (float*)smem;
    char*  akb  = smem + OFF_AK;
    char*  avb  = smem + OFF_AV;
    char*  wkb  = smem + OFF_WK;
    char*  wvb  = smem + OFF_WV;

    const int b   = blockIdx.x >> 6;
    const int p0  = (blockIdx.x & 63) * 64;
    const int tid = threadIdx.x;

    // ---- stage ge tile transposed into ldsG[p][c] -------------------------
    // Global side coalesced along positions; transpose happens in the LDS
    // write address.  cg is wave-uniform (cg = wave>>1), p_off is per-lane.
    {
        const int p_off = tid & 63;
        const int cg    = tid >> 6;            // 0..7, uniform per wave
        #pragma unroll 4
        for (int cb = 0; cb < 32; ++cb) {
            const int c = cb * 8 + cg;
#if USE_ASYNC_LDS
            const float* srow = ge + ((size_t)b * CC + c) * NP + p0;  // wave-uniform
            const unsigned va  = (unsigned)(p_off * 4);               // per-lane global byte off
            const unsigned ldo = (unsigned)((p_off * 257 + c) * 4);   // per-lane LDS byte off
            asm volatile("global_load_async_to_lds_b32 %0, %1, %2"
                         :: "v"(ldo), "v"(va), "s"(srow)
                         : "memory");
#else
            ldsG[p_off * 257 + c] = ge[((size_t)b * CC + c) * NP + p0 + p_off];
#endif
        }
    }
    // ---- stage weights into fragment-major bf16 layout --------------------
    for (int it = 0; it < 64; ++it) {
        const int e = it * 512 + tid;          // e = c*128 + j
        const int c = e >> 7, j = e & 127;
        const unsigned so = b_swz(c, j);
        *(unsigned short*)(wkb + so) = f2bf(wk[e]);
        *(unsigned short*)(wvb + so) = f2bf(wv[e]);
    }
#if USE_ASYNC_LDS
    asm volatile("s_wait_asynccnt 0" ::: "memory");
#endif
    __syncthreads();

    // ---- guiding + A-operand staging (coalesced along channels) -----------
    for (int it = 0; it < 32; ++it) {
        const int e = it * 512 + tid;          // 64 rows * 256 c
        const int p = e >> 8, c = e & 255;
        const size_t gidx = ((size_t)b * NP + p0 + p) * CC + c;
        const float g = ldsG[p * 257 + c] * keys[gidx];
        guiding_out[gidx] = g;
        const unsigned so = a_swz(p, c);
        *(unsigned short*)(avb + so) = f2bf(g);
        *(unsigned short*)(akb + so) = f2bf(g + key_pe[gidx]);
    }
    __syncthreads();

    // ---- WMMA: 16 waves; wave -> {matrix(k/v), prow(0..3), jhalf(0..1)} ----
    const int wave = tid >> 5, lane = tid & 31;
    const int mat   = wave >> 3;               // 0 = K, 1 = V
    const int prow  = (wave >> 1) & 3;
    const int jhalf = wave & 1;

    const char*  abase = mat ? avb : akb;
    const char*  wbase = mat ? wvb : wkb;
    const float* bias  = mat ? bv  : bk;
    unsigned short* outw = mat ? v_ws : k_ws;

    v8f acc[4];
    #pragma unroll
    for (int jt = 0; jt < 4; ++jt)
        #pragma unroll
        for (int r = 0; r < 8; ++r) acc[jt][r] = 0.0f;

    #pragma unroll
    for (int s = 0; s < 8; ++s) {              // K = 256 in 8 steps of 32
        FragBF a;
        const char* ap = abase + ((((prow * 8 + s) * 32) + lane) << 5);
        a.q[0] = *(const uint4v*)ap;
        a.q[1] = *(const uint4v*)(ap + 16);
        #pragma unroll
        for (int jt = 0; jt < 4; ++jt) {
            const int jtile = jhalf * 4 + jt;
            FragBF bb;
            const char* bp = wbase + ((((jtile * 8 + s) * 32) + lane) << 5);
            bb.q[0] = *(const uint4v*)bp;
            bb.q[1] = *(const uint4v*)(bp + 16);
            acc[jt] = __builtin_amdgcn_wmma_f32_16x16x32_bf16(
                false, a.v, false, bb.v, (short)0, acc[jt], false, false);
        }
    }

    // ---- store D tiles (+bias) as bf16 ------------------------------------
    const int nB = lane & 15;
    const int mD = (lane < 16) ? 0 : 8;
    #pragma unroll
    for (int jt = 0; jt < 4; ++jt) {
        const int j    = (jhalf * 4 + jt) * 16 + nB;
        const float bj = bias[j];
        #pragma unroll
        for (int r = 0; r < 8; ++r) {
            const int p = p0 + prow * 16 + mD + r;
            outw[((size_t)b * NP + p) * INT_ + j] = f2bf(acc[jt][r] + bj);
        }
    }
}

// ---------------------------------------------------------------------------
// K2: flash attention.  One WG per (b, head, q-row); 8 waves split 4096 keys.
// q projected in-kernel (1x16 per WG).  softmax scale 0.25 folded into q.
// ---------------------------------------------------------------------------
__global__ __launch_bounds__(256)
void k2_attention(const float* __restrict__ queries, const float* __restrict__ query_pe,
                  const float* __restrict__ wq, const float* __restrict__ bq,
                  const unsigned short* __restrict__ k_ws,
                  const unsigned short* __restrict__ v_ws,
                  float* __restrict__ attn_o)           // (B,8,128) f32
{
    __shared__ float part[16][17];
    __shared__ float qv[16];
    __shared__ float redm[256], redl[256], reda[256][16];
    __shared__ float Mg, Lg;

    const int b  = blockIdx.x >> 6;
    const int h  = (blockIdx.x >> 3) & 7;
    const int qr = blockIdx.x & 7;
    const int t  = threadIdx.x;

    {
        const int d = t & 15, seg = t >> 4;
        float s = 0.f;
        #pragma unroll 4
        for (int i = 0; i < 16; ++i) {
            const int c = seg * 16 + i;
            const size_t qi = ((size_t)b * TT + qr) * CC + c;
            s += (queries[qi] + query_pe[qi]) * wq[c * INT_ + h * DH + d];
        }
        part[seg][d] = s;
    }
    __syncthreads();
    if (t < 16) {
        float s = bq[h * DH + t];
        #pragma unroll
        for (int seg = 0; seg < 16; ++seg) s += part[seg][t];
        qv[t] = s * 0.25f;                     // 1/sqrt(16)
    }
    __syncthreads();

    float qvr[16];
    #pragma unroll
    for (int d = 0; d < 16; ++d) qvr[d] = qv[d];

    const int wave = t >> 5, lane = t & 31;
    float m = -1e30f, l = 0.f;
    float a[16];
    #pragma unroll
    for (int d = 0; d < 16; ++d) a[d] = 0.f;

    for (int i = 0; i < 16; ++i) {
        const int key = wave * 512 + i * 32 + lane;
        const unsigned short* kp = k_ws + ((size_t)b * NP + key) * INT_ + h * DH;
        const unsigned short* vp = v_ws + ((size_t)b * NP + key) * INT_ + h * DH;
        uint4v k0 = *(const uint4v*)kp;
        uint4v k1 = *(const uint4v*)(kp + 8);
        uint4v v0 = *(const uint4v*)vp;
        uint4v v1 = *(const uint4v*)(vp + 8);
        float kf[16], vf[16];
        #pragma unroll
        for (int d = 0; d < 4; ++d) {
            kf[2*d]   = bf_lo(k0[d]); kf[2*d+1]   = bf_hi(k0[d]);
            kf[8+2*d] = bf_lo(k1[d]); kf[8+2*d+1] = bf_hi(k1[d]);
            vf[2*d]   = bf_lo(v0[d]); vf[2*d+1]   = bf_hi(v0[d]);
            vf[8+2*d] = bf_lo(v1[d]); vf[8+2*d+1] = bf_hi(v1[d]);
        }

        float s = 0.f;
        #pragma unroll
        for (int d = 0; d < 16; ++d) s += qvr[d] * kf[d];

        const float nm    = fmaxf(m, s);
        const float scale = __expf(m - nm);
        const float p     = __expf(s - nm);
        l = l * scale + p;
        #pragma unroll
        for (int d = 0; d < 16; ++d) a[d] = a[d] * scale + p * vf[d];
        m = nm;
    }

    redm[t] = m; redl[t] = l;
    #pragma unroll
    for (int d = 0; d < 16; ++d) reda[t][d] = a[d];
    __syncthreads();

    if (t == 0) {
        float M = -1e30f;
        for (int i = 0; i < 256; ++i) M = fmaxf(M, redm[i]);
        Mg = M;
    }
    __syncthreads();

    float outd = 0.f;
    if (t < 16) {
        for (int i = 0; i < 256; ++i) outd += reda[i][t] * __expf(redm[i] - Mg);
    } else if (t == 16) {
        float L = 0.f;
        for (int i = 0; i < 256; ++i) L += redl[i] * __expf(redm[i] - Mg);
        Lg = L;
    }
    __syncthreads();
    if (t < 16)
        attn_o[((size_t)b * TT + qr) * INT_ + h * DH + t] = outd / Lg;
}

// ---------------------------------------------------------------------------
// K3: wo projection -> q2, both MLPs, token & query/pe assembly. One WG/batch.
// ---------------------------------------------------------------------------
__global__ __launch_bounds__(256)
void k3_tokens(const float* __restrict__ attn_o,
               const float* __restrict__ wo, const float* __restrict__ bo,
               const float* __restrict__ static_unc, const float* __restrict__ static_ref,
               const float* __restrict__ uw1, const float* __restrict__ ub1,
               const float* __restrict__ uw2, const float* __restrict__ ub2,
               const float* __restrict__ uw3, const float* __restrict__ ub3,
               const float* __restrict__ rw1, const float* __restrict__ rb1,
               const float* __restrict__ rw2, const float* __restrict__ rb2,
               const float* __restrict__ rw3, const float* __restrict__ rb3,
               const float* __restrict__ query_pe,
               float* __restrict__ q_out,        // (B,16,256)
               float* __restrict__ qpe_out,      // (B,16,256)
               float* __restrict__ utok_ws, float* __restrict__ rtok_ws,
               float* __restrict__ ctok_ws)
{
    __shared__ float ov[8][128];
    __shared__ float q2[8][256];
    __shared__ float h1[64], h2[64];

    const int b = blockIdx.x, t = threadIdx.x;

    #pragma unroll
    for (int i = 0; i < 4; ++i) {
        const int e = i * 256 + t;
        ov[e >> 7][e & 127] = attn_o[(size_t)b * 1024 + e];
    }
    __syncthreads();

    {   // q2 = out @ wo + bo
        const int c = t;
        #pragma unroll
        for (int r = 0; r < 8; ++r) {
            float s = bo[c];
            for (int k = 0; k < 128; ++k) s += ov[r][k] * wo[k * CC + c];
            q2[r][c] = s;
        }
    }
    __syncthreads();
    ctok_ws[b * CC + t] = q2[1][t];

    for (int which = 0; which < 2; ++which) {
        const float* sv = which ? static_ref : static_unc;
        const float* w1 = which ? rw1 : uw1; const float* b1 = which ? rb1 : ub1;
        const float* w2 = which ? rw2 : uw2; const float* b2 = which ? rb2 : ub2;
        const float* w3 = which ? rw3 : uw3; const float* b3 = which ? rb3 : ub3;

        if (t < 64) {
            float s = b1[t];
            for (int i = 0; i < 512; ++i) {
                const float xi = (i < 256) ? sv[i] : q2[1][i - 256];
                s += xi * w1[i * HID + t];
            }
            h1[t] = fmaxf(s, 0.f);
        }
        __syncthreads();
        if (t < 64) {
            float s = b2[t];
            for (int j = 0; j < 64; ++j) s += h1[j] * w2[j * HID + t];
            h2[t] = fmaxf(s, 0.f);
        }
        __syncthreads();
        float tokv = b3[t];
        for (int j = 0; j < 64; ++j) tokv += h2[j] * w3[j * CC + t];
        if (which == 0) {
            utok_ws[b * CC + t] = tokv;
        } else {
            rtok_ws[b * CC + t] = tokv;
            q_out[((size_t)b * 16 + 1) * CC + t] = tokv;   // q3 row 1 = ref_tok
        }
        __syncthreads();
    }

    #pragma unroll
    for (int r = 0; r < 8; ++r) {
        if (r != 1) q_out[((size_t)b * 16 + r) * CC + t] = q2[r][t];
        qpe_out[((size_t)b * 16 + r) * CC + t] = query_pe[((size_t)b * TT + r) * CC + t];
    }
}

// ---------------------------------------------------------------------------
// K4: masks.  One WG = (batch, 32-position tile); guiding read exactly once.
// ---------------------------------------------------------------------------
__global__ __launch_bounds__(256)
void k4_masks(const float* __restrict__ guiding,
              const float* __restrict__ utok_ws, const float* __restrict__ rtok_ws,
              const float* __restrict__ ctok_ws,
              float* __restrict__ unc_out, float* __restrict__ ref_out,
              float* __restrict__ coarse_out, float* __restrict__ final_out,
              float* __restrict__ ref_map_ws)
{
    __shared__ float gt[32][257];
    __shared__ float ut[256], rt[256], ct[256];
    __shared__ float red[32][8][3];

    const int b  = blockIdx.x >> 7;
    const int p0 = (blockIdx.x & 127) * 32;
    const int t  = threadIdx.x;

    ut[t] = utok_ws[b * CC + t];
    rt[t] = rtok_ws[b * CC + t];
    ct[t] = ctok_ws[b * CC + t];
    const float* gtile = guiding + ((size_t)b * NP + p0) * CC;
    for (int it = 0; it < 32; ++it) {
        const int e = it * 256 + t;
        gt[e >> 8][e & 255] = gtile[e];
        if (it < 31) __builtin_prefetch(gtile + e + 256, 0, 0);
    }
    __syncthreads();

    const int p = t >> 3, seg = t & 7;
    float su = 0.f, sr = 0.f, sc = 0.f;
    #pragma unroll 4
    for (int i = 0; i < 32; ++i) {
        const int c = seg * 32 + i;
        const float g = gt[p][c];
        su += g * ut[c]; sr += g * rt[c]; sc += g * ct[c];
    }
    red[p][seg][0] = su; red[p][seg][1] = sr; red[p][seg][2] = sc;
    __syncthreads();

    if (t < 32) {
        float u = 0.f, r = 0.f, c2 = 0.f;
        #pragma unroll
        for (int s = 0; s < 8; ++s) { u += red[t][s][0]; r += red[t][s][1]; c2 += red[t][s][2]; }
        const size_t o = (size_t)b * NP + p0 + t;
        const float un = 1.f / (1.f + __expf(-u));
        unc_out[o] = un; ref_out[o] = r; coarse_out[o] = c2;
        final_out[o] = (un >= 0.5f) ? r : c2;
        const float sg_r = 1.f / (1.f + __expf(-r));
        const float sg_c = 1.f / (1.f + __expf(-c2));
        ref_map_ws[o] = un * (sg_r - sg_c);
    }
}

// ---------------------------------------------------------------------------
// K5: top-4 max and top-4 min of ref_map per batch (indices sorted ascending).
// ---------------------------------------------------------------------------
__global__ __launch_bounds__(256)
void k5_topk(const float* __restrict__ ref_map_ws, int* __restrict__ idx_ws)
{
    __shared__ float vals[4096];
    __shared__ float rv[256];
    __shared__ int   ri[256];
    __shared__ int   chosen[8];

    const int b = blockIdx.x, t = threadIdx.x;
    for (int i = 0; i < 16; ++i) vals[i * 256 + t] = ref_map_ws[(size_t)b * NP + i * 256 + t];
    __syncthreads();

    for (int phase = 0; phase < 2; ++phase) {          // 0 = max picks, 1 = min picks
        if (phase == 1) {
            __syncthreads();
            for (int i = 0; i < 16; ++i) vals[i * 256 + t] = ref_map_ws[(size_t)b * NP + i * 256 + t];
            __syncthreads();
        }
        for (int sel = 0; sel < 4; ++sel) {
            float best = phase ? 1e30f : -1e30f; int bi = 0;
            for (int i = 0; i < 16; ++i) {
                const int idx = t * 16 + i;
                const float v = vals[idx];
                const bool better = phase ? (v < best) : (v > best);
                if (better) { best = v; bi = idx; }
            }
            rv[t] = best; ri[t] = bi;
            __syncthreads();
            for (int s = 128; s > 0; s >>= 1) {
                if (t < s) {
                    const float a0 = rv[t], a1 = rv[t + s];
                    const bool better = phase ? (a1 < a0) : (a1 > a0);
                    if (better || (a1 == a0 && ri[t + s] < ri[t])) { rv[t] = a1; ri[t] = ri[t + s]; }
                }
                __syncthreads();
            }
            if (t == 0) {
                const int w = ri[0];
                chosen[phase * 4 + sel] = w;
                vals[w] = phase ? 1e30f : -1e30f;       // exclude from later picks
            }
            __syncthreads();
        }
    }
    if (t == 0) {
        for (int g = 0; g < 2; ++g) {
            int* a = &chosen[g * 4];
            for (int i = 0; i < 4; ++i)
                for (int j = i + 1; j < 4; ++j)
                    if (a[j] < a[i]) { int tmp = a[i]; a[i] = a[j]; a[j] = tmp; }
            for (int i = 0; i < 4; ++i) idx_ws[b * 8 + g * 4 + i] = a[i];
        }
    }
}

// ---------------------------------------------------------------------------
// K6: gather sampled points: PE encodings + content = keys*(1+ge).
// ---------------------------------------------------------------------------
__global__ __launch_bounds__(256)
void k6_gather(const float* __restrict__ keys, const float* __restrict__ ge,
               const float* __restrict__ pe_gauss,
               const float* __restrict__ pt_pos, const float* __restrict__ pt_neg,
               const int* __restrict__ idx_ws,
               float* __restrict__ q_out, float* __restrict__ qpe_out)
{
    const int b = blockIdx.x, t = threadIdx.x;
    for (int s = 0; s < 8; ++s) {
        const int idx = idx_ws[b * 8 + s];
        const float* pt = (s < 4) ? pt_pos : pt_neg;
        const int row = 8 + s;
        if (t < 128) {
            const int xc = idx & 63, yr = idx >> 6;
            const float cx = 2.f * ((xc + 0.5f) / 64.f) - 1.f;
            const float cy = 2.f * ((yr + 0.5f) / 64.f) - 1.f;
            const float f = 6.283185307179586f * (cx * pe_gauss[t] + cy * pe_gauss[128 + t]);
            qpe_out[((size_t)b * 16 + row) * CC + t]       = __sinf(f) + pt[t];
            qpe_out[((size_t)b * 16 + row) * CC + 128 + t] = __cosf(f) + pt[128 + t];
        }
        const float g = ge[((size_t)b * CC + t) * NP + idx];
        const float k = keys[((size_t)b * NP + idx) * CC + t];
        q_out[((size_t)b * 16 + row) * CC + t] = k * (1.f + g);
    }
}

// ---------------------------------------------------------------------------
// Launch
// ---------------------------------------------------------------------------
extern "C" void kernel_launch(void* const* d_in, const int* in_sizes, int n_in,
                              void* d_out, int out_size, void* d_ws, size_t ws_size,
                              hipStream_t stream) {
    const float* queries    = (const float*)d_in[0];
    const float* keys       = (const float*)d_in[1];
    const float* query_pe   = (const float*)d_in[2];
    const float* key_pe     = (const float*)d_in[3];
    const float* ge         = (const float*)d_in[6];   // guiding_embedding (B,C,64,64)
    const float* wq = (const float*)d_in[8];  const float* bq = (const float*)d_in[9];
    const float* wk = (const float*)d_in[10]; const float* bk = (const float*)d_in[11];
    const float* wv = (const float*)d_in[12]; const float* bv = (const float*)d_in[13];
    const float* wo = (const float*)d_in[14]; const float* bo = (const float*)d_in[15];
    const float* static_unc = (const float*)d_in[16];
    const float* static_ref = (const float*)d_in[17];
    const float* uw1 = (const float*)d_in[18]; const float* ub1 = (const float*)d_in[19];
    const float* uw2 = (const float*)d_in[20]; const float* ub2 = (const float*)d_in[21];
    const float* uw3 = (const float*)d_in[22]; const float* ub3 = (const float*)d_in[23];
    const float* rw1 = (const float*)d_in[24]; const float* rb1 = (const float*)d_in[25];
    const float* rw2 = (const float*)d_in[26]; const float* rb2 = (const float*)d_in[27];
    const float* rw3 = (const float*)d_in[28]; const float* rb3 = (const float*)d_in[29];
    const float* pe_gauss = (const float*)d_in[30];
    const float* pt_pos   = (const float*)d_in[31];
    const float* pt_neg   = (const float*)d_in[32];
    // d_in[33] == n_sample_points == 4 (hardcoded)

    float* out = (float*)d_out;
    float* o_queries = out;                         // (32,16,256)   131072
    float* o_guiding = out + 131072;                // (32,4096,256) 33554432
    float* o_qpe     = out + 33685504;              // (32,16,256)
    float* o_unc     = out + 33816576;              // (32,1,64,64)
    float* o_refm    = out + 33947648;
    float* o_coarse  = out + 34078720;
    float* o_final   = out + 34209792;

    // workspace partition (~67.8 MB)
    char* ws = (char*)d_ws;
    unsigned short* k_ws = (unsigned short*)ws;                 ws += (size_t)BB * NP * INT_ * 2;
    unsigned short* v_ws = (unsigned short*)ws;                 ws += (size_t)BB * NP * INT_ * 2;
    float* attn_o  = (float*)ws;                                ws += (size_t)BB * TT * INT_ * 4;
    float* utok_ws = (float*)ws;                                ws += (size_t)BB * CC * 4;
    float* rtok_ws = (float*)ws;                                ws += (size_t)BB * CC * 4;
    float* ctok_ws = (float*)ws;                                ws += (size_t)BB * CC * 4;
    float* ref_map = (float*)ws;                                ws += (size_t)BB * NP * 4;
    int*   idx_ws  = (int*)ws;                                  ws += (size_t)BB * 8 * 4;

    hipFuncSetAttribute((const void*)k1_guiding_kvproj,
                        hipFuncAttributeMaxDynamicSharedMemorySize, K1_LDS);

    k1_guiding_kvproj<<<BB * 64, 512, K1_LDS, stream>>>(
        ge, keys, key_pe, wk, bk, wv, bv, o_guiding, k_ws, v_ws);

    k2_attention<<<BB * NH * TT, 256, 0, stream>>>(
        queries, query_pe, wq, bq, k_ws, v_ws, attn_o);

    k3_tokens<<<BB, 256, 0, stream>>>(
        attn_o, wo, bo, static_unc, static_ref,
        uw1, ub1, uw2, ub2, uw3, ub3, rw1, rb1, rw2, rb2, rw3, rb3,
        query_pe, o_queries, o_qpe, utok_ws, rtok_ws, ctok_ws);

    k4_masks<<<BB * 128, 256, 0, stream>>>(
        o_guiding, utok_ws, rtok_ws, ctok_ws,
        o_unc, o_refm, o_coarse, o_final, ref_map);

    k5_topk<<<BB, 256, 0, stream>>>(ref_map, idx_ws);

    k6_gather<<<BB, 256, 0, stream>>>(
        keys, ge, pe_gauss, pt_pos, pt_neg, idx_ws, o_queries, o_qpe);
}
